// WordEmbeddingNewsEncoder_25769804123
// MI455X (gfx1250) — matrix operand order repaired and checked by reading, not compiled, not run on hardware
//
#include <hip/hip_runtime.h>
#include <hip/hip_bf16.h>

typedef __attribute__((ext_vector_type(16))) _Float16 v16h;
typedef __attribute__((ext_vector_type(8)))  _Float16 v8h;
typedef __attribute__((ext_vector_type(8)))  float    v8f;

#define NBATCH 4096
#define LLEN   32
#define DDIM   300
#define DPAD   320
#define NHEAD  6
#define HDIM   50
#define HDPAD  64
#define WSTRIDE (DPAD*DPAD)   // 102400 halfs per padded weight matrix

// ---- LDS layout (bytes) ----
#define OFF_XS    0        // [32][320] f16 : x, later ctx16 (pre-Wo ctx, f16)
#define OFF_QHS   20480    // [6][32][64] f16 : per-head Q, later co16 [32][320] f16
#define OFF_KHS   45056    // [6][32][64] f16 : per-head K, later co32 [32][320] f32 (spills into VHST)
#define OFF_VHST  69632    // [6][64][32] f16 : per-head V transposed
#define OFF_SS    94208    // [6][32][32] f32 : attention scores
#define OFF_WS    118784   // [6][32][32] f16 : softmax weights
#define OFF_AS    131072   // [32] f32 : additive-attention logits
#define OFF_ALPHA 131200   // [32] f32 : pooling alphas
#define SMEM_BYTES 131328

__device__ __forceinline__ v8f wmma16(v16h a, v16h b, v8f c) {
  // D = A(16x32 f16) * B(32x16 f16) + C(16x16 f32)
  return __builtin_amdgcn_wmma_f32_16x16x32_f16(false, a, false, b, (short)0, c,
                                                false, false);
}

// A fragment: row-major f16 matrix, row stride ld (halfs).
// Lane layout (16-bit A 16x32): lanes 0-15 -> K 0..15, lanes 16-31 -> K 16..31.
__device__ __forceinline__ v16h load_A(const _Float16* src, int ld, int m0, int k0, int lane) {
  int m  = m0 + (lane & 15);
  int ka = k0 + ((lane & 16) ? 16 : 0);
  v8h lo = *(const v8h*)(src + m * ld + ka);
  v8h hi = *(const v8h*)(src + m * ld + ka + 8);
  return __builtin_shufflevector(lo, hi, 0,1,2,3,4,5,6,7,8,9,10,11,12,13,14,15);
}

// B fragment: B[k][n] = Arr[n][k] (Arr row-major, row stride ld halfs).
// Lane layout (16-bit B 32x16): lanes 0-15 col n, K 0..15; lanes 16-31 col n, K 16..31.
__device__ __forceinline__ v16h load_Brow(const _Float16* arr, int ld, int n0, int k0, int lane) {
  int n = n0 + (lane & 15);
  int k = k0 + ((lane & 16) ? 16 : 0);
  return *(const v16h*)(arr + n * ld + k);
}

// ---------------- weight pre-conversion: f32 [300x300] -> padded f16 [320x320] ----------------
__global__ void cvt_weights(const float* __restrict__ Wq, const float* __restrict__ Wk,
                            const float* __restrict__ Wv, const float* __restrict__ Wo,
                            const float* __restrict__ Va, _Float16* __restrict__ w16) {
  int idx = blockIdx.x * blockDim.x + threadIdx.x;
  if (idx >= 5 * WSTRIDE) return;
  int which = idx / WSTRIDE;
  int rem   = idx % WSTRIDE;
  int j = rem / DPAD, d = rem % DPAD;
  const float* W = (which == 0) ? Wq : (which == 1) ? Wk : (which == 2) ? Wv
                 : (which == 3) ? Wo : Va;
  float v = (j < DDIM && d < DDIM) ? W[j * DDIM + d] : 0.0f;
  w16[idx] = (_Float16)v;
}

// ---------------- fused news encoder: one block (8 wave32) per batch element ----------------
__global__ __launch_bounds__(256)
void news_encoder(const int* __restrict__ title, const float* __restrict__ emb,
                  const float* __restrict__ pos, const _Float16* __restrict__ w16,
                  const float* __restrict__ bq, const float* __restrict__ bk,
                  const float* __restrict__ bv, const float* __restrict__ bo,
                  const float* __restrict__ ba, const float* __restrict__ qw,
                  float* __restrict__ out) {
  __shared__ __align__(32) char smem[SMEM_BYTES];
  const int tid  = threadIdx.x;
  const int lane = tid & 31;
  const int wv   = tid >> 5;
  const int b    = blockIdx.x;

  _Float16* xs    = (_Float16*)(smem + OFF_XS);
  _Float16* qhs   = (_Float16*)(smem + OFF_QHS);
  _Float16* khs   = (_Float16*)(smem + OFF_KHS);
  _Float16* vhsT  = (_Float16*)(smem + OFF_VHST);
  float*    ss    = (float*)   (smem + OFF_SS);
  _Float16* wsf16 = (_Float16*)(smem + OFF_WS);
  _Float16* ctx16 = (_Float16*)(smem + OFF_XS);    // alias: x dead after QKV
  _Float16* co16  = (_Float16*)(smem + OFF_QHS);   // alias: qhs dead after scores
  float*    co32  = (float*)   (smem + OFF_KHS);   // alias: khs/vhsT dead after ctx
  float*    asum  = (float*)   (smem + OFF_AS);
  float*    alph  = (float*)   (smem + OFF_ALPHA);

  // ---- phase 0: embedding gather + positional add, store f16 padded; zero head buffers ----
  for (int i = tid; i < LLEN * DPAD; i += 256) {
    int l = i / DPAD, d = i % DPAD;
    float v = 0.0f;
    if (d < DDIM) {
      int tok = title[b * LLEN + l];
      v = emb[tok * DDIM + d] + pos[l * DDIM + d];
    }
    xs[i] = (_Float16)v;
  }
  for (int i = tid; i < 3 * NHEAD * LLEN * HDPAD; i += 256)
    ((_Float16*)(smem + OFF_QHS))[i] = (_Float16)0.0f;   // qhs,khs,vhsT contiguous
  __syncthreads();

  // ---- phase 1: Q,K,V projections. Work unit = (matrix, m-tile, n-group-of-4):
  //      one A fragment feeds 4 WMMAs into 4 accumulators (A reuse x4). ----
  const float rs = 0.14142135623730951f;  // 1/sqrt(HD)
  for (int g = wv; g < 30; g += 8) {
    int which = g / 10, rem = g % 10;
    int mt = rem / 5, nt0 = (rem % 5) * 4;
    const _Float16* Wsel = w16 + which * WSTRIDE;
    v8f c0 = {}, c1 = {}, c2 = {}, c3 = {};
#pragma unroll
    for (int kb = 0; kb < 10; ++kb) {
      v16h a  = load_A(xs, DPAD, mt * 16, kb * 32, lane);
      v16h b0 = load_Brow(Wsel, DPAD, (nt0 + 0) * 16, kb * 32, lane);
      v16h b1 = load_Brow(Wsel, DPAD, (nt0 + 1) * 16, kb * 32, lane);
      v16h b2 = load_Brow(Wsel, DPAD, (nt0 + 2) * 16, kb * 32, lane);
      v16h b3 = load_Brow(Wsel, DPAD, (nt0 + 3) * 16, kb * 32, lane);
      c0 = wmma16(a, b0, c0);
      c1 = wmma16(a, b1, c1);
      c2 = wmma16(a, b2, c2);
      c3 = wmma16(a, b3, c3);
    }
    const float* bias = (which == 0) ? bq : (which == 1) ? bk : bv;
#pragma unroll
    for (int j = 0; j < 4; ++j) {
      v8f c = (j == 0) ? c0 : (j == 1) ? c1 : (j == 2) ? c2 : c3;
      int n = (nt0 + j) * 16 + (lane & 15);
      if (n < DDIM) {
        float bz = bias[n];
        int h = n / HDIM, cc = n - h * HDIM;
#pragma unroll
        for (int r = 0; r < 8; ++r) {
          int m = mt * 16 + r + ((lane & 16) ? 8 : 0);
          float vvv = c[r] + bz;
          if (which == 0)      qhs[h * LLEN * HDPAD + m * HDPAD + cc] = (_Float16)(vvv * rs);
          else if (which == 1) khs[h * LLEN * HDPAD + m * HDPAD + cc] = (_Float16)vvv;
          else                 vhsT[h * HDPAD * LLEN + cc * LLEN + m] = (_Float16)vvv;
        }
      }
    }
  }
  __syncthreads();

  // ---- phase 2: scores[h] = q_h [32x64] x k_h^T -> [32x32].
  //      Work unit = (head, m-tile), 2 n-tiles per unit, 2 K-steps. ----
  for (int g = wv; g < 12; g += 8) {
    int h = g >> 1, mt = g & 1;
    const _Float16* qh = qhs + h * LLEN * HDPAD;
    const _Float16* kh = khs + h * LLEN * HDPAD;
    v8f c0 = {}, c1 = {};
#pragma unroll
    for (int kb = 0; kb < 2; ++kb) {
      v16h a  = load_A(qh, HDPAD, mt * 16, kb * 32, lane);
      v16h b0 = load_Brow(kh, HDPAD, 0,  kb * 32, lane);
      v16h b1 = load_Brow(kh, HDPAD, 16, kb * 32, lane);
      c0 = wmma16(a, b0, c0);
      c1 = wmma16(a, b1, c1);
    }
#pragma unroll
    for (int j = 0; j < 2; ++j) {
      v8f c = (j == 0) ? c0 : c1;
      int n = j * 16 + (lane & 15);
#pragma unroll
      for (int r = 0; r < 8; ++r) {
        int m = mt * 16 + r + ((lane & 16) ? 8 : 0);
        ss[h * LLEN * LLEN + m * LLEN + n] = c[r];
      }
    }
  }
  __syncthreads();

  // ---- phase 3: row softmax over scores -> f16 weights; spare threads zero pads ----
  if (tid < NHEAD * LLEN) {
    int h = tid >> 5, row = tid & 31;
    const float* sr = ss + h * LLEN * LLEN + row * LLEN;
    float mx = -1e30f;
    for (int j = 0; j < LLEN; ++j) mx = fmaxf(mx, sr[j]);
    float sum = 0.0f;
    for (int j = 0; j < LLEN; ++j) sum += __expf(sr[j] - mx);
    float inv = 1.0f / sum;
    _Float16* wr = wsf16 + h * LLEN * LLEN + row * LLEN;
    for (int j = 0; j < LLEN; ++j) wr[j] = (_Float16)(__expf(sr[j] - mx) * inv);
  } else {
    int i = tid - NHEAD * LLEN;
    if (i < 32) asum[i] = 0.0f;
    for (int j = i; j < LLEN * (DPAD - DDIM); j += 64) {  // zero ctx16 pad columns
      int l = j / (DPAD - DDIM), d = DDIM + j % (DPAD - DDIM);
      ctx16[l * DPAD + d] = (_Float16)0.0f;
    }
  }
  __syncthreads();

  // ---- phase 4: ctx[h] = softmax_w [32x32] x v_h [32x50pad64].
  //      Work unit = (head, m-tile): 1 A load feeds 4 WMMAs (one K-step). ----
  for (int g = wv; g < 12; g += 8) {
    int h = g >> 1, mt = g & 1;
    const _Float16* wh = wsf16 + h * LLEN * LLEN;
    const _Float16* vh = vhsT + h * HDPAD * LLEN;
    v16h a  = load_A(wh, LLEN, mt * 16, 0, lane);
    v16h b0 = load_Brow(vh, LLEN, 0,  0, lane);
    v16h b1 = load_Brow(vh, LLEN, 16, 0, lane);
    v16h b2 = load_Brow(vh, LLEN, 32, 0, lane);
    v16h b3 = load_Brow(vh, LLEN, 48, 0, lane);
    v8f c0 = {}, c1 = {}, c2 = {}, c3 = {};
    c0 = wmma16(a, b0, c0);
    c1 = wmma16(a, b1, c1);
    c2 = wmma16(a, b2, c2);
    c3 = wmma16(a, b3, c3);
#pragma unroll
    for (int j = 0; j < 4; ++j) {
      v8f c = (j == 0) ? c0 : (j == 1) ? c1 : (j == 2) ? c2 : c3;
      int cc = j * 16 + (lane & 15);
      if (cc < HDIM) {
        int dcol = h * HDIM + cc;
#pragma unroll
        for (int r = 0; r < 8; ++r) {
          int m = mt * 16 + r + ((lane & 16) ? 8 : 0);
          ctx16[m * DPAD + dcol] = (_Float16)c[r];
        }
      }
    }
  }
  __syncthreads();

  // ---- phase 5: output projection co = ctx @ Wo.T + bo (4-wide N-blocking) ----
  {
    const _Float16* WoW = w16 + 3 * WSTRIDE;
    for (int g = wv; g < 10; g += 8) {
      int mt = g / 5, nt0 = (g % 5) * 4;
      v8f c0 = {}, c1 = {}, c2 = {}, c3 = {};
#pragma unroll
      for (int kb = 0; kb < 10; ++kb) {
        v16h a  = load_A(ctx16, DPAD, mt * 16, kb * 32, lane);
        v16h b0 = load_Brow(WoW, DPAD, (nt0 + 0) * 16, kb * 32, lane);
        v16h b1 = load_Brow(WoW, DPAD, (nt0 + 1) * 16, kb * 32, lane);
        v16h b2 = load_Brow(WoW, DPAD, (nt0 + 2) * 16, kb * 32, lane);
        v16h b3 = load_Brow(WoW, DPAD, (nt0 + 3) * 16, kb * 32, lane);
        c0 = wmma16(a, b0, c0);
        c1 = wmma16(a, b1, c1);
        c2 = wmma16(a, b2, c2);
        c3 = wmma16(a, b3, c3);
      }
#pragma unroll
      for (int j = 0; j < 4; ++j) {
        v8f c = (j == 0) ? c0 : (j == 1) ? c1 : (j == 2) ? c2 : c3;
        int n = (nt0 + j) * 16 + (lane & 15);
        float bz = (n < DDIM) ? bo[n] : 0.0f;
#pragma unroll
        for (int r = 0; r < 8; ++r) {
          int m = mt * 16 + r + ((lane & 16) ? 8 : 0);
          float vvv = (n < DDIM) ? (c[r] + bz) : 0.0f;
          co32[m * DPAD + n] = vvv;
          co16[m * DPAD + n] = (_Float16)vvv;
        }
      }
    }
  }
  __syncthreads();

  // ---- phase 6: additive attention logits a[l] = sum_d tanh(co@Va.T+ba)[l,d]*qw[d] ----
  {
    const _Float16* VaW = w16 + 4 * WSTRIDE;
    for (int g = wv; g < 10; g += 8) {
      int mt = g / 5, nt0 = (g % 5) * 4;
      v8f c0 = {}, c1 = {}, c2 = {}, c3 = {};
#pragma unroll
      for (int kb = 0; kb < 10; ++kb) {
        v16h a  = load_A(co16, DPAD, mt * 16, kb * 32, lane);
        v16h b0 = load_Brow(VaW, DPAD, (nt0 + 0) * 16, kb * 32, lane);
        v16h b1 = load_Brow(VaW, DPAD, (nt0 + 1) * 16, kb * 32, lane);
        v16h b2 = load_Brow(VaW, DPAD, (nt0 + 2) * 16, kb * 32, lane);
        v16h b3 = load_Brow(VaW, DPAD, (nt0 + 3) * 16, kb * 32, lane);
        c0 = wmma16(a, b0, c0);
        c1 = wmma16(a, b1, c1);
        c2 = wmma16(a, b2, c2);
        c3 = wmma16(a, b3, c3);
      }
#pragma unroll
      for (int j = 0; j < 4; ++j) {
        v8f c = (j == 0) ? c0 : (j == 1) ? c1 : (j == 2) ? c2 : c3;
        int n = (nt0 + j) * 16 + (lane & 15);
        if (n < DDIM) {
          float bz = ba[n], qn = qw[n];
#pragma unroll
          for (int r = 0; r < 8; ++r) {
            int m = mt * 16 + r + ((lane & 16) ? 8 : 0);
            float tv = tanhf(c[r] + bz);
            atomicAdd(&asum[m], tv * qn);
          }
        }
      }
    }
  }
  __syncthreads();

  // ---- phase 7: softmax over 32 logits (wave 0, shuffle reductions) ----
  if (wv == 0) {
    float av = asum[lane];
    float mx = av;
#pragma unroll
    for (int off = 16; off > 0; off >>= 1) mx = fmaxf(mx, __shfl_xor(mx, off, 32));
    float e = __expf(av - mx);
    float s = e;
#pragma unroll
    for (int off = 16; off > 0; off >>= 1) s += __shfl_xor(s, off, 32);
    alph[lane] = e / s;
  }
  __syncthreads();

  // ---- phase 8: out[b,d] = sum_l alpha[l] * co32[l,d] ----
  for (int d = tid; d < DDIM; d += 256) {
    float acc = 0.0f;
#pragma unroll 4
    for (int l = 0; l < LLEN; ++l) acc += alph[l] * co32[l * DPAD + d];
    out[b * DDIM + d] = acc;
  }
}

extern "C" void kernel_launch(void* const* d_in, const int* in_sizes, int n_in,
                              void* d_out, int out_size, void* d_ws, size_t ws_size,
                              hipStream_t stream) {
  (void)in_sizes; (void)n_in; (void)out_size; (void)ws_size;
  const int*   title = (const int*)  d_in[0];
  const float* emb   = (const float*)d_in[1];
  const float* pos   = (const float*)d_in[2];
  const float* Wq    = (const float*)d_in[3];
  const float* bq    = (const float*)d_in[4];
  const float* Wk    = (const float*)d_in[5];
  const float* bk    = (const float*)d_in[6];
  const float* Wv    = (const float*)d_in[7];
  const float* bv    = (const float*)d_in[8];
  const float* Wo    = (const float*)d_in[9];
  const float* bo    = (const float*)d_in[10];
  const float* Va    = (const float*)d_in[11];
  const float* ba    = (const float*)d_in[12];
  const float* qw    = (const float*)d_in[13];
  float* out = (float*)d_out;
  _Float16* w16 = (_Float16*)d_ws;   // 5 * 320*320 * 2B = 1,024,000 bytes

  int cvt_total = 5 * WSTRIDE;
  cvt_weights<<<(cvt_total + 255) / 256, 256, 0, stream>>>(Wq, Wk, Wv, Wo, Va, w16);
  news_encoder<<<NBATCH, 256, 0, stream>>>(title, emb, pos, w16, bq, bk, bv, bo,
                                           ba, qw, out);
}
